// Block_15298673508693
// MI455X (gfx1250) — matrix operand order repaired and checked by reading, not compiled, not run on hardware
//
#include <hip/hip_runtime.h>
#include <hip/hip_bf16.h>
#include <cstdint>

// ---------------------------------------------------------------------------
// Types for CDNA5 WMMA (wave32, 16x16x32 bf16 -> f32)
// ---------------------------------------------------------------------------
typedef __attribute__((ext_vector_type(16))) __bf16 bf16x16;
typedef __attribute__((ext_vector_type(8)))  __bf16 bf16x8;
typedef __attribute__((ext_vector_type(8)))  float  f32x8;

#define WMMA_BF16(A, B, C) \
  __builtin_amdgcn_wmma_f32_16x16x32_bf16(false, (A), false, (B), (short)0, (C), false, false)

// ---------------------------------------------------------------------------
// gfx1250 async global->LDS copy (ASYNCcnt path), 16 bytes per lane.
// ISA: GLOBAL_LOAD_ASYNC_TO_LDS_B128 vdst(LDS byte offset), vaddr(64b), off
// ---------------------------------------------------------------------------
__device__ inline void async_copy16(const void* g, void* l) {
  uint32_t loff = (uint32_t)(uintptr_t)(__attribute__((address_space(3))) void*)l;
  uint64_t ga   = (uint64_t)(uintptr_t)g;
  asm volatile("global_load_async_to_lds_b128 %0, %1, off"
               :: "v"(loff), "v"(ga) : "memory");
}

template <int N> __device__ inline void wait_asynccnt() {
#if __has_builtin(__builtin_amdgcn_s_wait_asynccnt)
  __builtin_amdgcn_s_wait_asynccnt(N);
#else
  asm volatile("s_wait_asynccnt %0" :: "n"(N) : "memory");
#endif
}

// A-fragment (16x32, bf16): lane r=lane%16 holds row r; kh=lane/16 selects K
// halves. elems 0..7 = K[kh*8 .. kh*8+7], elems 8..15 = K[16+kh*8 ..].
__device__ inline bf16x16 frag_a(const __bf16* base, int stride) {
  const int lane = threadIdx.x & 31;
  const int r = lane & 15, kh = lane >> 4;
  const __bf16* p = base + r * stride + kh * 8;
  bf16x8 lo = *(const bf16x8*)(p);
  bf16x8 hi = *(const bf16x8*)(p + 16);
  bf16x16 f;
#pragma unroll
  for (int i = 0; i < 8; ++i) { f[i] = lo[i]; f[i + 8] = hi[i]; }
  return f;
}

// B-fragment (32x16) loaded from W in row-major [N][K] form (B = W^T):
// lane c=lane%16 holds output column c; kh=lane/16: elems 0..7 = K[kh*8 .. kh*8+7], elems 8..15 = K[16+kh*8 ..].
__device__ inline bf16x16 frag_b(const __bf16* base, int stride) {
  const int lane = threadIdx.x & 31;
  const int c = lane & 15, kh = lane >> 4;
  const __bf16* p = base + c * stride + kh * 8;
  bf16x8 lo = *(const bf16x8*)(p);
  bf16x8 hi = *(const bf16x8*)(p + 16);
  bf16x16 f;
#pragma unroll
  for (int i = 0; i < 8; ++i) { f[i] = lo[i]; f[i + 8] = hi[i]; }
  return f;
}

// ---------------------------------------------------------------------------
// Elementwise fp32 -> bf16 (weights)
// ---------------------------------------------------------------------------
__global__ void cvt_bf16_kernel(const float* __restrict__ s, __bf16* __restrict__ d, int n) {
  int i = blockIdx.x * 256 + threadIdx.x;
  if (i < n) d[i] = (__bf16)s[i];
}

// ---------------------------------------------------------------------------
// LayerNorm over C=768, one block per row, output bf16
// ---------------------------------------------------------------------------
__global__ __launch_bounds__(256) void ln_kernel(const float* __restrict__ x,
                                                 const float* __restrict__ w,
                                                 const float* __restrict__ b,
                                                 __bf16* __restrict__ out) {
  const int row = blockIdx.x;
  const float* xr = x + (size_t)row * 768;
  float loc[3];
  float s = 0.f, ss = 0.f;
#pragma unroll
  for (int j = 0; j < 3; ++j) {
    float t = xr[threadIdx.x + 256 * j];
    loc[j] = t; s += t; ss += t * t;
  }
#pragma unroll
  for (int o = 16; o >= 1; o >>= 1) {
    s  += __shfl_xor(s,  o, 32);
    ss += __shfl_xor(ss, o, 32);
  }
  __shared__ float red[16];
  const int wv = threadIdx.x >> 5, ln = threadIdx.x & 31;
  if (ln == 0) { red[wv] = s; red[8 + wv] = ss; }
  __syncthreads();
  float S = 0.f, SS = 0.f;
#pragma unroll
  for (int i = 0; i < 8; ++i) { S += red[i]; SS += red[8 + i]; }
  const float mu  = S * (1.f / 768.f);
  const float var = SS * (1.f / 768.f) - mu * mu;
  const float inv = rsqrtf(var + 1e-5f);
#pragma unroll
  for (int j = 0; j < 3; ++j) {
    int c = threadIdx.x + 256 * j;
    out[(size_t)row * 768 + c] = (__bf16)((loc[j] - mu) * inv * w[c] + b[c]);
  }
}

// ---------------------------------------------------------------------------
// Positional attention scores: pos[h][i][j] = softmax_j(rel(i,j) . pos_w[h] + pos_b[h])
// Batch-independent; computed once. Grid = H*196 (one row per block).
// ---------------------------------------------------------------------------
__global__ __launch_bounds__(256) void pos_kernel(const float* __restrict__ pw,
                                                  const float* __restrict__ pb,
                                                  float* __restrict__ pos) {
  const int h = blockIdx.x / 196, i = blockIdx.x % 196;
  const int iy = i / 14, ix = i % 14;
  const int t = threadIdx.x;
  __shared__ float red[256];
  float raw = 0.f, val = -1e30f;
  if (t < 196) {
    int jy = t / 14, jx = t % 14;
    float dx = (float)(jx - ix), dy = (float)(jy - iy);
    raw = dx * pw[h * 3 + 0] + dy * pw[h * 3 + 1] + (dx * dx + dy * dy) * pw[h * 3 + 2] + pb[h];
    val = raw;
  }
  red[t] = val; __syncthreads();
  for (int o = 128; o > 0; o >>= 1) { if (t < o) red[t] = fmaxf(red[t], red[t + o]); __syncthreads(); }
  const float mx = red[0]; __syncthreads();
  float ex = (t < 196) ? expf(raw - mx) : 0.f;
  red[t] = ex; __syncthreads();
  for (int o = 128; o > 0; o >>= 1) { if (t < o) red[t] += red[t + o]; __syncthreads(); }
  const float sm = red[0];
  if (t < 196) pos[((size_t)h * 196 + i) * 196 + t] = ex / sm;
}

// ---------------------------------------------------------------------------
// bf16 GEMM: out[M,N] = A[M,K] * W[N,K]^T (+epilogue).
// Block tile 128x128, 8 waves (4x2), wave tile 32x64 (2x4 WMMA accums), BK=32.
// Double-buffered async global->LDS pipeline (ASYNCcnt).
// EPI: 0 = store bf16; 1 = +bias, exact GELU, store bf16; 2 = +bias, +resid, store f32.
// ---------------------------------------------------------------------------
template <int EPI>
__global__ __launch_bounds__(256, 2) void gemm_bf16(const __bf16* __restrict__ A,
                                                    const __bf16* __restrict__ W,
                                                    const float* __restrict__ bias,
                                                    const float* __restrict__ resid,
                                                    void* __restrict__ outv,
                                                    int K, int Ndim) {
  alignas(16) __shared__ __bf16 As[2][128 * 40];
  alignas(16) __shared__ __bf16 Ws[2][128 * 40];
  const int tid  = threadIdx.x;
  const int wave = tid >> 5, lane = tid & 31;
  const int wm = wave >> 1, wn = wave & 1;
  const size_t rowBase = (size_t)blockIdx.y * 128;
  const size_t colBase = (size_t)blockIdx.x * 128;

  f32x8 acc[2][4];
#pragma unroll
  for (int mi = 0; mi < 2; ++mi)
#pragma unroll
    for (int ni = 0; ni < 4; ++ni) acc[mi][ni] = (f32x8){};

  const int crow = tid >> 1, cseg = tid & 1;  // each thread copies 16 bf16 of one row
  const __bf16* aSrc = A + (rowBase + crow) * (size_t)K + cseg * 16;
  const __bf16* wSrc = W + (colBase + crow) * (size_t)K + cseg * 16;
  const int dstOff = crow * 40 + cseg * 16;

  // stage 0 prologue
  async_copy16(aSrc,     &As[0][dstOff]);
  async_copy16(aSrc + 8, &As[0][dstOff + 8]);
  async_copy16(wSrc,     &Ws[0][dstOff]);
  async_copy16(wSrc + 8, &Ws[0][dstOff + 8]);

  const int nIter = K >> 5;
  for (int it = 0; it < nIter; ++it) {
    const int stage = it & 1;
    if (it + 1 < nIter) {
      const int k1 = (it + 1) * 32;
      async_copy16(aSrc + k1,     &As[stage ^ 1][dstOff]);
      async_copy16(aSrc + k1 + 8, &As[stage ^ 1][dstOff + 8]);
      async_copy16(wSrc + k1,     &Ws[stage ^ 1][dstOff]);
      async_copy16(wSrc + k1 + 8, &Ws[stage ^ 1][dstOff + 8]);
      wait_asynccnt<4>();   // current stage's 4 copies retired; next 4 in flight
    } else {
      wait_asynccnt<0>();
    }
    __syncthreads();        // all waves' copies for `stage` visible

    bf16x16 af[2], bf_[4];
#pragma unroll
    for (int mi = 0; mi < 2; ++mi) af[mi] = frag_a(&As[stage][(wm * 32 + mi * 16) * 40], 40);
#pragma unroll
    for (int ni = 0; ni < 4; ++ni) bf_[ni] = frag_b(&Ws[stage][(wn * 64 + ni * 16) * 40], 40);
#pragma unroll
    for (int mi = 0; mi < 2; ++mi)
#pragma unroll
      for (int ni = 0; ni < 4; ++ni)
        acc[mi][ni] = WMMA_BF16(af[mi], bf_[ni], acc[mi][ni]);

    __syncthreads();        // compute done before this buffer is refilled
  }

  const int cn = lane & 15, kh = lane >> 4;
#pragma unroll
  for (int mi = 0; mi < 2; ++mi)
#pragma unroll
    for (int ni = 0; ni < 4; ++ni) {
      const size_t gc = colBase + wn * 64 + ni * 16 + cn;
#pragma unroll
      for (int vg = 0; vg < 8; ++vg) {
        const size_t gr = rowBase + wm * 32 + mi * 16 + vg + 8 * kh;
        float v = acc[mi][ni][vg];
        if constexpr (EPI >= 1) v += bias[gc];
        if constexpr (EPI == 1) v = 0.5f * v * (1.f + erff(v * 0.70710678118f));
        const size_t oi = gr * (size_t)Ndim + gc;
        if constexpr (EPI == 2) {
          ((float*)outv)[oi] = v + resid[oi];
        } else {
          ((__bf16*)outv)[oi] = (__bf16)v;
        }
      }
    }
}

// ---------------------------------------------------------------------------
// GPSA attention: one block per (b,h). N=196 (padded 208/224), d=48 (padded 64).
// attn = (1-sig)*softmax(q k^T * scale) + sig*pos  (already row-normalized).
// ---------------------------------------------------------------------------
__global__ __launch_bounds__(256, 1) void attn_kernel(const __bf16* __restrict__ qk,
                                                      const __bf16* __restrict__ vmat,
                                                      const float* __restrict__ pos,
                                                      const float* __restrict__ gating,
                                                      __bf16* __restrict__ outp) {
  const int bh = blockIdx.x, b = bh >> 4, h = bh & 15;
  alignas(16) __shared__ __bf16 sh_k[208 * 64];    // keys [n][d], zero padded
  alignas(16) __shared__ __bf16 sh_vt[48 * 224];   // V^T  [d][n], zero padded
  alignas(16) __shared__ __bf16 sh_q[16 * 64];     // q m-tile
  alignas(16) __shared__ float  sh_s[16 * 224];    // scores f32, reused as bf16 probs

  const int tid = threadIdx.x, wave = tid >> 5, lane = tid & 31;
  const __bf16* qb = qk + (size_t)b * 196 * 1536 + h * 48;
  const __bf16* kb = qk + (size_t)b * 196 * 1536 + 768 + h * 48;
  const __bf16* vb = vmat + (size_t)b * 196 * 768 + h * 48;

  for (int idx = tid; idx < 208 * 64; idx += 256) {
    int n = idx >> 6, c = idx & 63;
    sh_k[idx] = (n < 196 && c < 48) ? kb[n * 1536 + c] : (__bf16)0.f;
  }
  for (int idx = tid; idx < 48 * 224; idx += 256) {
    int dd = idx / 224, n = idx % 224;
    sh_vt[idx] = (n < 196) ? vb[n * 768 + dd] : (__bf16)0.f;
  }
  const float sig   = 1.f / (1.f + expf(-gating[h]));
  const float scale = rsqrtf(48.f);

  for (int mt = 0; mt < 13; ++mt) {
    for (int idx = tid; idx < 16 * 64; idx += 256) {
      int r = idx >> 6, c = idx & 63, m = mt * 16 + r;
      sh_q[idx] = (m < 196 && c < 48) ? qb[m * 1536 + c] : (__bf16)0.f;
    }
    __syncthreads();  // q ready; previous P*V done before sh_s overwrite

    // S = q k^T over 13 n-tiles (208 cols); waves own tiles {w, w+8}
#pragma unroll
    for (int tsel = 0; tsel < 2; ++tsel) {
      const int nt = wave + tsel * 8;
      if (nt < 13) {
        f32x8 acc = (f32x8){};
#pragma unroll
        for (int ks = 0; ks < 64; ks += 32) {
          bf16x16 a  = frag_a(sh_q + ks, 64);
          bf16x16 bb = frag_b(sh_k + nt * 16 * 64 + ks, 64);
          acc = WMMA_BF16(a, bb, acc);
        }
        const int cn = lane & 15, kh = lane >> 4;
#pragma unroll
        for (int vg = 0; vg < 8; ++vg)
          sh_s[(vg + 8 * kh) * 224 + nt * 16 + cn] = acc[vg];
      }
    }
    __syncthreads();

    // softmax + mix with pos: 16 threads per row; shuffle-reduce within groups
    const int r = tid >> 4, g = tid & 15, m = mt * 16 + r;
    float vals[14];
#pragma unroll
    for (int j = 0; j < 14; ++j) {
      int c = g + 16 * j;
      vals[j] = (c < 208) ? sh_s[r * 224 + c] : 0.f;
    }
    float mx = -1e30f;
#pragma unroll
    for (int j = 0; j < 14; ++j) { int c = g + 16 * j; if (c < 196) mx = fmaxf(mx, vals[j] * scale); }
#pragma unroll
    for (int o = 8; o >= 1; o >>= 1) mx = fmaxf(mx, __shfl_xor(mx, o, 16));
    float sm = 0.f;
#pragma unroll
    for (int j = 0; j < 14; ++j) { int c = g + 16 * j; if (c < 196) sm += expf(vals[j] * scale - mx); }
#pragma unroll
    for (int o = 8; o >= 1; o >>= 1) sm += __shfl_xor(sm, o, 16);
    const float inv = 1.f / sm;
    __syncthreads();  // all f32 reads complete before in-place bf16 writes

    __bf16* sb = (__bf16*)sh_s;
#pragma unroll
    for (int j = 0; j < 14; ++j) {
      int c = g + 16 * j;
      float p = 0.f;
      if (c < 196 && m < 196) {
        float patch = expf(vals[j] * scale - mx) * inv;
        float po = pos[((size_t)h * 196 + m) * 196 + c];
        p = (1.f - sig) * patch + sig * po;  // row sums to 1 -> no renorm needed
      }
      sb[r * 224 + c] = (__bf16)p;
    }
    __syncthreads();

    // out = P * V : 3 d-tiles of 16, K = 224 (7 WMMA steps)
    if (wave < 3) {
      f32x8 acc = (f32x8){};
#pragma unroll
      for (int kk = 0; kk < 224; kk += 32) {
        bf16x16 a  = frag_a(sb + kk, 224);
        bf16x16 bb = frag_b(sh_vt + wave * 16 * 224 + kk, 224);
        acc = WMMA_BF16(a, bb, acc);
      }
      const int cn = lane & 15, kh = lane >> 4;
#pragma unroll
      for (int vg = 0; vg < 8; ++vg) {
        int mrow = mt * 16 + vg + 8 * kh;
        if (mrow < 196)
          outp[((size_t)b * 196 + mrow) * 768 + h * 48 + wave * 16 + cn] = (__bf16)acc[vg];
      }
    }
    __syncthreads();
  }
}

// ---------------------------------------------------------------------------
// Orchestration
// ---------------------------------------------------------------------------
extern "C" void kernel_launch(void* const* d_in, const int* in_sizes, int n_in,
                              void* d_out, int out_size, void* d_ws, size_t ws_size,
                              hipStream_t stream) {
  (void)in_sizes; (void)n_in; (void)out_size; (void)ws_size;
  const float* x      = (const float*)d_in[0];
  const float* n1w    = (const float*)d_in[1];
  const float* n1b    = (const float*)d_in[2];
  const float* qk_w   = (const float*)d_in[3];
  const float* v_w    = (const float*)d_in[4];
  const float* pos_w  = (const float*)d_in[5];
  const float* pos_b  = (const float*)d_in[6];
  const float* gating = (const float*)d_in[7];
  const float* proj_w = (const float*)d_in[8];
  const float* proj_b = (const float*)d_in[9];
  const float* n2w    = (const float*)d_in[10];
  const float* n2b    = (const float*)d_in[11];
  const float* fc1_w  = (const float*)d_in[12];
  const float* fc1_b  = (const float*)d_in[13];
  const float* fc2_w  = (const float*)d_in[14];
  const float* fc2_b  = (const float*)d_in[15];

  const int M = 64 * 196;  // 12544 = 98 * 128
  char* ws = (char*)d_ws;
  __bf16* xn   = (__bf16*)(ws + 0);            // 19267584 B
  __bf16* qkb  = (__bf16*)(ws + 19267584);     // 38535168 B
  __bf16* vbuf = (__bf16*)(ws + 57802752);     // 19267584 B
  __bf16* hdn  = (__bf16*)(ws + 0);            // reuses xn|qkb|vbuf (77070336 B)
  __bf16* atn  = (__bf16*)(ws + 77070336);     // 19267584 B (reused as xn2)
  __bf16* xn2  = atn;
  float*  x1   = (float*) (ws + 96337920);     // 38535168 B
  float*  pos  = (float*) (ws + 134873088);    // 2458624 B
  __bf16* wq   = (__bf16*)(ws + 137331712);    // 2359296 B
  __bf16* wv   = (__bf16*)(ws + 139691008);    // 1179648 B
  __bf16* wp   = (__bf16*)(ws + 140870656);    // 1179648 B
  __bf16* w1   = (__bf16*)(ws + 142050304);    // 4718592 B
  __bf16* w2   = (__bf16*)(ws + 146768896);    // 4718592 B

  // weight conversions
  cvt_bf16_kernel<<<(1179648 + 255) / 256, 256, 0, stream>>>(qk_w, wq, 1179648);
  cvt_bf16_kernel<<<(589824 + 255) / 256, 256, 0, stream>>>(v_w, wv, 589824);
  cvt_bf16_kernel<<<(589824 + 255) / 256, 256, 0, stream>>>(proj_w, wp, 589824);
  cvt_bf16_kernel<<<(2359296 + 255) / 256, 256, 0, stream>>>(fc1_w, w1, 2359296);
  cvt_bf16_kernel<<<(2359296 + 255) / 256, 256, 0, stream>>>(fc2_w, w2, 2359296);

  // LN1, pos scores
  ln_kernel<<<M, 256, 0, stream>>>(x, n1w, n1b, xn);
  pos_kernel<<<16 * 196, 256, 0, stream>>>(pos_w, pos_b, pos);

  // qk = xn @ qk_w^T ; v = xn @ v_w^T  (bf16 stores)
  gemm_bf16<0><<<dim3(12, 98), 256, 0, stream>>>(xn, wq, nullptr, nullptr, qkb, 768, 1536);
  gemm_bf16<0><<<dim3(6, 98), 256, 0, stream>>>(xn, wv, nullptr, nullptr, vbuf, 768, 768);

  // attention
  attn_kernel<<<64 * 16, 256, 0, stream>>>(qkb, vbuf, pos, gating, atn);

  // x1 = x + attn_out @ proj_w^T + proj_b
  gemm_bf16<2><<<dim3(6, 98), 256, 0, stream>>>(atn, wp, proj_b, x, (void*)x1, 768, 768);

  // MLP branch
  ln_kernel<<<M, 256, 0, stream>>>(x1, n2w, n2b, xn2);
  gemm_bf16<1><<<dim3(24, 98), 256, 0, stream>>>(xn2, w1, fc1_b, nullptr, hdn, 768, 3072);
  gemm_bf16<2><<<dim3(6, 98), 256, 0, stream>>>(hdn, w2, fc2_b, x1, d_out, 3072, 768);
}
